// GraphAttentionLayer_77919296684441
// MI455X (gfx1250) — matrix operand order repaired and checked by reading, not compiled, not run on hardware
//
#include <hip/hip_runtime.h>
#include <hip/hip_bf16.h>

#define Bsz 8
#define Nn  2048
#define Ff  128
#define ALPHA 0.2f
#define LN_EPS 1e-5f
#define NEG_INF (-9.0e15f)

typedef __attribute__((ext_vector_type(16))) __bf16 v16bf;
typedef __attribute__((ext_vector_type(8)))  float  v8f;

union BfVec { v16bf v; uint4 q[2]; };

// -------------------------------------------------------------------------
// Kernel 1: h = LN((x*(nw+1)+nb) @ W);  s1 = h@a1; s2 = h@a2
// grid (N/16, B), 256 threads (8 waves). Wave w computes 16x16 tile of h
// at cols [16w,16w+16) with 4 x v_wmma_f32_16x16x32_bf16 (K=128).
// -------------------------------------------------------------------------
__global__ __launch_bounds__(256) void k1_gemm_ln(
    const float* __restrict__ x, const float* __restrict__ W,
    const float* __restrict__ a, const float* __restrict__ nw,
    const float* __restrict__ nb, const float* __restrict__ gamma,
    const float* __restrict__ beta,
    float* __restrict__ h, __bf16* __restrict__ hT,
    float* __restrict__ s1, float* __restrict__ s2)
{
  __shared__ __align__(16) __bf16 ldsX[16][Ff];    // xhat tile (bf16)  4KB
  __shared__ __align__(16) __bf16 ldsWT[Ff][Ff];   // W^T (bf16)       32KB
  __shared__ float  hTile[16][Ff];                 // h tile (f32)      8KB
  __shared__ float  mu_s[16], rs_s[16];
  __shared__ float  red1[256], red2[256];

  const int t  = threadIdx.x;
  const int it = blockIdx.x;
  const int b  = blockIdx.y;
  const int r0 = it * 16;
  const int lane = t & 31;
  const int w    = t >> 5;

  // stage W^T as bf16 (coalesced reads, transposed LDS stores)
  for (int i = 0; i < 64; ++i) {
    int idx = i * 256 + t;           // linear over row-major W
    int kr  = idx >> 7;              // k (row of W)
    int c   = idx & 127;             // output feature
    ldsWT[c][kr] = (__bf16)W[idx];
  }
  // stage xhat tile (16 x 128)
  {
    int r  = t >> 4;
    int cb = (t & 15) * 8;
    int n  = r0 + r;
    float sc = nw[n] + 1.0f;
    float bi = nb[n];
    const float* xr = x + ((size_t)b * Nn + n) * Ff + cb;
    for (int i = 0; i < 8; ++i)
      ldsX[r][cb + i] = (__bf16)(xr[i] * sc + bi);
  }
  __syncthreads();

  // per-wave WMMA GEMM
  {
    const int row = lane & 15;
    const int hs  = lane >> 4;
    const int col = w * 16 + (lane & 15);
    v8f acc = {};
    for (int s = 0; s < 4; ++s) {
      BfVec au, bu;
      // A 16x32: elems 0..7 -> K = s*32 + hs*8 + e ; elems 8..15 -> +16
      au.q[0] = *(const uint4*)&ldsX[row][s*32 + hs*8];
      au.q[1] = *(const uint4*)&ldsX[row][s*32 + 16 + hs*8];
      // B 32x16: elems 0..7 -> K = s*32 + hs*8 + e ; elems 8..15 -> +16, from W^T[col][K]
      bu.q[0] = *(const uint4*)&ldsWT[col][s*32 + hs*8];
      bu.q[1] = *(const uint4*)&ldsWT[col][s*32 + 16 + hs*8];
      acc = __builtin_amdgcn_wmma_f32_16x16x32_bf16(
          false, au.v, false, bu.v, (short)0, acc, false, false);
    }
    for (int v = 0; v < 8; ++v)
      hTile[v + hs*8][col] = acc[v];
  }
  __syncthreads();

  // LayerNorm statistics (one thread per row; cheap at this size)
  if (t < 16) {
    float m = 0.f;
    for (int c = 0; c < Ff; ++c) m += hTile[t][c];
    m *= (1.0f / Ff);
    float vv = 0.f;
    for (int c = 0; c < Ff; ++c) { float d = hTile[t][c] - m; vv += d * d; }
    vv *= (1.0f / Ff);
    mu_s[t] = m;
    rs_s[t] = rsqrtf(vv + LN_EPS);
  }
  __syncthreads();

  // normalize, emit h (f32 row-major) + h^T (bf16 F-major), s1/s2 partials
  {
    int r  = t >> 4;
    int cb = (t & 15) * 8;
    float m = mu_s[r], rs = rs_s[r];
    float p1 = 0.f, p2 = 0.f;
    int n = r0 + r;
    float* hrow = h + ((size_t)b * Nn + n) * Ff;
    for (int i = 0; i < 8; ++i) {
      int c = cb + i;
      float hv = (hTile[r][c] - m) * rs * gamma[c] + beta[c];
      hrow[c] = hv;
      hT[((size_t)b * Ff + c) * Nn + n] = (__bf16)hv;
      p1 += hv * a[c];
      p2 += hv * a[Ff + c];
    }
    red1[t] = p1; red2[t] = p2;
  }
  __syncthreads();
  if (t < 16) {
    float q1 = 0.f, q2 = 0.f;
    for (int i = 0; i < 16; ++i) { q1 += red1[t*16+i]; q2 += red2[t*16+i]; }
    s1[(size_t)b * Nn + r0 + t] = q1;
    s2[(size_t)b * Nn + r0 + t] = q2;
  }
}

// -------------------------------------------------------------------------
// Kernel 2: flash-style masked softmax + attn@h + residual + elu.
// grid (N/16, B), 256 threads. 16 rows per block; rank-1 scores
// e_ij = lrelu(s1_i + s2_j) masked by adj never touch HBM.
// Phase C: K=128 chunks, double-buffered prob tile in LDS ->
//          1 barrier per 4 WMMAs instead of 2 barriers per WMMA.
// -------------------------------------------------------------------------
__global__ __launch_bounds__(256) void k2_attn(
    const int* __restrict__ adj,
    const float* __restrict__ h, const __bf16* __restrict__ hT,
    const float* __restrict__ s1g, const float* __restrict__ s2g,
    float* __restrict__ out)
{
  __shared__ float s2_s[Nn];                           // 8KB
  __shared__ float s1_s[16];
  __shared__ float m_s[16], is_s[16];
  __shared__ float red[256];
  __shared__ __align__(16) __bf16 P_s[2][16][128];     // prob tile, 2x4KB

  const int t  = threadIdx.x;
  const int it = blockIdx.x;
  const int b  = blockIdx.y;
  const int r0 = it * 16;
  const int lane = t & 31;
  const int w    = t >> 5;

  for (int i = t; i < Nn; i += 256) s2_s[i] = s2g[(size_t)b * Nn + i];
  if (t < 16) s1_s[t] = s1g[(size_t)b * Nn + r0 + t];
  __syncthreads();

  const int   r    = t >> 4;        // row within slab (16 threads per row)
  const int   c16  = t & 15;
  const int*  arow = adj + (size_t)(r0 + r) * Nn;
  const float s1v  = s1_s[r];

  // phase A: row max of masked leaky-relu scores
  float mx = NEG_INF;
  for (int j = c16; j < Nn; j += 16) {
    __builtin_prefetch(arow + j + 256, 0, 1);
    float e = s1v + s2_s[j];
    e = e > 0.f ? e : ALPHA * e;
    e = arow[j] > 0 ? e : NEG_INF;
    mx = fmaxf(mx, e);
  }
  red[t] = mx;
  __syncthreads();
  if (t < 16) {
    float m = red[t*16];
    for (int i = 1; i < 16; ++i) m = fmaxf(m, red[t*16+i]);
    m_s[t] = m;
  }
  __syncthreads();
  const float mrow = m_s[r];

  // phase B: row sum of exp
  float sm = 0.f;
  for (int j = c16; j < Nn; j += 16) {
    float e = s1v + s2_s[j];
    e = e > 0.f ? e : ALPHA * e;
    e = arow[j] > 0 ? e : NEG_INF;
    sm += __expf(e - mrow);
  }
  red[t] = sm;
  __syncthreads();
  if (t < 16) {
    float q = 0.f;
    for (int i = 0; i < 16; ++i) q += red[t*16+i];
    is_s[t] = 1.0f / q;
  }
  __syncthreads();
  const float inv = is_s[r];

  // phase C: P(16x128,bf16) @ h(128x16,bf16), f32 accumulate, K = 2048
  const int row = lane & 15;
  const int hs  = lane >> 4;
  const int col = w * 16 + (lane & 15);
  const __bf16* hTb = hT + ((size_t)b * Ff + col) * Nn;   // h^T[col][*]
  v8f acc = {};
  int buf = 0;
  for (int j0 = 0; j0 < Nn; j0 += 128, buf ^= 1) {
    // 256 threads fill the 16x128 probability tile: 8 contiguous each
    {
      int kk = c16 * 8;
      const int*   aj  = arow + j0 + kk;
      const float* s2p = &s2_s[j0 + kk];
      __bf16*      pd  = &P_s[buf][r][kk];
      __builtin_prefetch(hTb + j0 + 128, 0, 1);
      for (int i = 0; i < 8; ++i) {
        float e = s1v + s2p[i];
        e = e > 0.f ? e : ALPHA * e;
        e = aj[i] > 0 ? e : NEG_INF;
        pd[i] = (__bf16)(__expf(e - mrow) * inv);
      }
    }
    __syncthreads();   // publish buf (also fences reads of buf from 2 iters ago)
    for (int s = 0; s < 4; ++s) {
      BfVec au, bu;
      au.q[0] = *(const uint4*)&P_s[buf][row][s*32 + hs*8];
      au.q[1] = *(const uint4*)&P_s[buf][row][s*32 + 16 + hs*8];
      bu.q[0] = *(const uint4*)&hTb[j0 + s*32 + hs*8];
      bu.q[1] = *(const uint4*)&hTb[j0 + s*32 + 16 + hs*8];
      acc = __builtin_amdgcn_wmma_f32_16x16x32_bf16(
          false, au.v, false, bu.v, (short)0, acc, false, false);
    }
  }

  // epilogue: + residual, elu
  const float* hbase = h   + ((size_t)b * Nn + r0) * Ff;
  float*       obase = out + ((size_t)b * Nn + r0) * Ff;
  for (int v = 0; v < 8; ++v) {
    int rr = v + hs * 8;
    float hp = acc[v] + hbase[rr * Ff + col];
    obase[rr * Ff + col] = hp > 0.f ? hp : (__expf(hp) - 1.0f);
  }
}

// -------------------------------------------------------------------------
extern "C" void kernel_launch(void* const* d_in, const int* in_sizes, int n_in,
                              void* d_out, int out_size, void* d_ws, size_t ws_size,
                              hipStream_t stream) {
  const float* x     = (const float*)d_in[0];
  const int*   adj   = (const int*)  d_in[1];
  const float* W     = (const float*)d_in[2];
  const float* a     = (const float*)d_in[3];
  const float* nw    = (const float*)d_in[4];
  const float* nb    = (const float*)d_in[5];
  const float* gamma = (const float*)d_in[6];
  const float* beta  = (const float*)d_in[7];
  float* out = (float*)d_out;

  char* ws = (char*)d_ws;
  const size_t H_BYTES  = (size_t)Bsz * Nn * Ff * sizeof(float);   // 8 MB
  const size_t HT_BYTES = (size_t)Bsz * Nn * Ff * sizeof(__bf16);  // 4 MB
  const size_t S_BYTES  = (size_t)Bsz * Nn * sizeof(float);        // 64 KB
  float*  h  = (float*)ws;
  __bf16* hT = (__bf16*)(ws + H_BYTES);
  float*  s1 = (float*)(ws + H_BYTES + HT_BYTES);
  float*  s2 = (float*)(ws + H_BYTES + HT_BYTES + S_BYTES);

  dim3 grid(Nn / 16, Bsz);
  k1_gemm_ln<<<grid, 256, 0, stream>>>(x, W, a, nw, nb, gamma, beta,
                                       h, hT, s1, s2);
  k2_attn<<<grid, 256, 0, stream>>>(adj, h, hT, s1, s2, out);
}
